// PNAFrozenCombined_5214090297743
// MI455X (gfx1250) — matrix operand order repaired and checked by reading, not compiled, not run on hardware
//
#include <hip/hip_runtime.h>
#include <stdint.h>

// ---------------------------------------------------------------------------
// PNA GNN (3DInfomax-style) for MI455X / gfx1250.
// All GEMMs run on v_wmma_f32_16x16x32_bf16 (wave32). Weights are re-packed
// into WMMA B-fragment order each launch; A tiles staged through LDS.
// ---------------------------------------------------------------------------

#define NN 50000
#define EE 400000
#define GG 2000
#define HH 128
#define NPG 25

typedef __bf16 bf16_t;
typedef bf16_t v16bf __attribute__((ext_vector_type(16)));
typedef float  v8f   __attribute__((ext_vector_type(8)));
typedef unsigned u32x4 __attribute__((ext_vector_type(4)));

union Frag16 { v16bf v; u32x4 q[2]; };

__device__ inline unsigned f2bfbits(float f) {
  unsigned x = __float_as_uint(f);
  unsigned r = x + 0x7FFFu + ((x >> 16) & 1u);
  return r >> 16;
}
__device__ inline unsigned packbf(float a, float b) {
  return f2bfbits(a) | (f2bfbits(b) << 16);
}
__device__ inline unsigned ordf(float f) {   // order-preserving f32 -> u32
  unsigned b = __float_as_uint(f);
  return (b & 0x80000000u) ? ~b : (b ^ 0x80000000u);
}
__device__ inline float unordf(unsigned u) {
  return __uint_as_float((u & 0x80000000u) ? (u ^ 0x80000000u) : ~u);
}
__device__ inline v8f wmma_bf16(v16bf a, v16bf b, v8f c) {
  return __builtin_amdgcn_wmma_f32_16x16x32_bf16(false, a, false, b,
                                                 (short)0, c, false, false);
}

// ---------------------------------------------------------------------------
// Small utility kernels
// ---------------------------------------------------------------------------
__global__ void k_zero_deg(unsigned* d) {
  int i = blockIdx.x * blockDim.x + threadIdx.x;
  if (i < NN) d[i] = 0u;
}
__global__ void k_count(const int* __restrict__ dst, unsigned* __restrict__ d) {
  int i = blockIdx.x * blockDim.x + threadIdx.x;
  if (i < EE) atomicAdd(d + dst[i], 1u);
}
__global__ void k_scalars(const unsigned* __restrict__ degU,
                          float* __restrict__ sdinv, float* __restrict__ amp,
                          float* __restrict__ att, float* __restrict__ hin) {
  int i = blockIdx.x * blockDim.x + threadIdx.x;
  if (i >= NN) return;
  float dg = (float)degU[i];
  sdinv[i] = 1.0f / fmaxf(dg, 1.0f);
  float ld = logf(dg + 1.0f);
  amp[i] = ld;                         // D_LOG == 1
  att[i] = (ld > 0.0f) ? (1.0f / ld) : 0.0f;
  hin[i] = (dg > 0.0f) ? 1.0f : 0.0f;
}
__global__ void k_init_aggr(float* __restrict__ s1, float* __restrict__ s2,
                            unsigned* __restrict__ mxu, unsigned* __restrict__ mnu) {
  int i = blockIdx.x * blockDim.x + threadIdx.x;
  if (i >= NN * HH) return;
  s1[i] = 0.0f; s2[i] = 0.0f; mxu[i] = 0u; mnu[i] = 0xFFFFFFFFu;
}

// Pack fp32 [K,N] weight into bf16 WMMA-B fragment order.
// packed[((kt*NT+nt)*32+lane)*16+e] = W[kt*32 + (e>>3)*16 + (lane>>4)*8 + (e&7)][nt*16 + (lane&15)]
__global__ void k_pack(const float* __restrict__ src, unsigned short* __restrict__ dst,
                       int K, int Ncols) {
  int id = blockIdx.x * blockDim.x + threadIdx.x;
  int total = K * Ncols;
  if (id >= total) return;
  int e = id & 15;
  int lane = (id >> 4) & 31;
  int frag = id >> 9;
  int NT = Ncols >> 4;
  int nt = frag % NT;
  int kt = frag / NT;
  int krow = kt * 32 + (e >> 3) * 16 + (lane >> 4) * 8 + (e & 7);
  int col = nt * 16 + (lane & 15);
  dst[id] = (unsigned short)f2bfbits(src[(size_t)krow * Ncols + col]);
}

// Node embedding: h[n][c] = sum_j aemb[nf[n][j] + 16*j][c]; writes f32 + bf16.
__global__ void k_embed(const int* __restrict__ nf, const float* __restrict__ emb,
                        float* __restrict__ h, unsigned short* __restrict__ hbf) {
  int id = blockIdx.x * blockDim.x + threadIdx.x;
  if (id >= NN * 64) return;
  int node = id >> 6;
  int c = (id & 63) * 2;
  const int* f = nf + node * 9;
  float a0 = 0.f, a1 = 0.f;
#pragma unroll
  for (int j = 0; j < 9; ++j) {
    const float* row = emb + (size_t)(f[j] + j * 16) * HH;
    a0 += row[c]; a1 += row[c + 1];
  }
  size_t idx = (size_t)node * HH + c;
  h[idx] = a0; h[idx + 1] = a1;
  reinterpret_cast<unsigned*>(hbf)[idx >> 1] = packbf(a0, a1);
}

// ---------------------------------------------------------------------------
// pretrans: z = [h[src], h[dst], e] @ w_pre + b_pre, scatter-reduce into
// s1 (sum), s2 (sumsq), mxu/mnu (ordered-u32 max/min) per dst node.
// Block = 128 threads (4 waves) = 64 edges; wave does 16 edges x 128 cols.
// ---------------------------------------------------------------------------
__global__ __launch_bounds__(128) void k_pretrans(
    const unsigned short* __restrict__ hbf, const float* __restrict__ bemb,
    const int* __restrict__ ef, const int* __restrict__ srcI,
    const int* __restrict__ dstI,
    const unsigned short* __restrict__ pw, const float* __restrict__ bias,
    float* __restrict__ s1, float* __restrict__ s2,
    unsigned* __restrict__ mxu, unsigned* __restrict__ mnu) {
  __shared__ __align__(16) unsigned short Ab[64 * 384];
  __shared__ float bembS[15 * HH];
  __shared__ int dstS[64];

  const int t = threadIdx.x;
  const int e0 = blockIdx.x * 64;

  for (int i = t; i < 15 * HH; i += 128) bembS[i] = bemb[i];
  if (t < 64) dstS[t] = dstI[e0 + t];
  __syncthreads();

  // Stage A rows: [hbf[src] | hbf[dst] | edge-embedding] as bf16
  for (int task = t; task < 192; task += 128) {
    int row = task / 3;
    int seg = task - row * 3;
    unsigned* q = reinterpret_cast<unsigned*>(Ab + row * 384);
    if (seg == 0) {
      const unsigned* p =
          reinterpret_cast<const unsigned*>(hbf + (size_t)srcI[e0 + row] * HH);
#pragma unroll 8
      for (int j = 0; j < 64; ++j) q[j] = p[j];
    } else if (seg == 1) {
      const unsigned* p =
          reinterpret_cast<const unsigned*>(hbf + (size_t)dstI[e0 + row] * HH);
#pragma unroll 8
      for (int j = 0; j < 64; ++j) q[64 + j] = p[j];
    } else {
      int eb = (e0 + row) * 3;
      const float* ba = bembS + (ef[eb] + 0) * HH;
      const float* bb = bembS + (ef[eb + 1] + 5) * HH;
      const float* bc = bembS + (ef[eb + 2] + 10) * HH;
#pragma unroll 4
      for (int c = 0; c < HH; c += 2) {
        float v0 = ba[c] + bb[c] + bc[c];
        float v1 = ba[c + 1] + bb[c + 1] + bc[c + 1];
        q[128 + (c >> 1)] = packbf(v0, v1);
      }
    }
  }
  __syncthreads();

  const int wv = t >> 5, lane = t & 31, hl = lane >> 4, ln = lane & 15;
  const int rowb = wv * 16;
  v8f acc[8];
#pragma unroll
  for (int i = 0; i < 8; ++i) acc[i] = (v8f){0, 0, 0, 0, 0, 0, 0, 0};

#pragma unroll 1
  for (int kt = 0; kt < 12; ++kt) {
    Frag16 a;
    const u32x4* ap = reinterpret_cast<const u32x4*>(
        Ab + (rowb + ln) * 384 + kt * 32 + hl * 8);
    a.q[0] = ap[0]; a.q[1] = ap[2];
#pragma unroll
    for (int nt = 0; nt < 8; ++nt) {
      Frag16 b;
      const u32x4* bp = reinterpret_cast<const u32x4*>(
          pw + ((size_t)(kt * 8 + nt)) * 512 + lane * 16);
      b.q[0] = bp[0]; b.q[1] = bp[1];
      acc[nt] = wmma_bf16(a.v, b.v, acc[nt]);
    }
  }

#pragma unroll
  for (int nt = 0; nt < 8; ++nt) {
    int col = nt * 16 + ln;
    float bc = bias[col];
#pragma unroll
    for (int r = 0; r < 8; ++r) {
      int m = hl * 8 + r;
      int dn = dstS[rowb + m];
      float z = acc[nt][r] + bc;
      size_t idx = (size_t)dn * HH + col;
      atomicAdd(s1 + idx, z);
      atomicAdd(s2 + idx, z * z);
      unsigned o = ordf(z);
      atomicMax(mxu + idx, o);
      atomicMin(mnu + idx, o);
    }
  }
}

// ---------------------------------------------------------------------------
// posttrans: h_new = [h, f(12H)] @ w_post + b_post + h ; also refresh hbf.
// Block = 128 threads / 64 nodes; K streamed as 13 chunks of 128.
// ---------------------------------------------------------------------------
__device__ inline float aggval(int qsel, const float* s1, const float* s2,
                               const unsigned* mxu, const unsigned* mnu,
                               size_t idx, float sdi) {
  if (qsel == 0) return s1[idx] * sdi;
  if (qsel == 1) return unordf(mxu[idx]);
  if (qsel == 2) return unordf(mnu[idx]);
  float mean = s1[idx] * sdi;
  return sqrtf(fmaxf(s2[idx] * sdi - mean * mean, 0.0f) + 1e-5f);
}

__global__ __launch_bounds__(128) void k_posttrans(
    float* __restrict__ h, unsigned short* __restrict__ hbf,
    const float* __restrict__ s1, const float* __restrict__ s2,
    const unsigned* __restrict__ mxu, const unsigned* __restrict__ mnu,
    const float* __restrict__ sdinv, const float* __restrict__ amp,
    const float* __restrict__ att, const float* __restrict__ hin,
    const unsigned short* __restrict__ pw, const float* __restrict__ bias) {
  __shared__ __align__(16) unsigned short Ab[64 * HH];
  const int t = threadIdx.x;
  const int n0 = blockIdx.x * 64;
  const int wv = t >> 5, lane = t & 31, hl = lane >> 4, ln = lane & 15;

  v8f acc[8];
#pragma unroll
  for (int i = 0; i < 8; ++i) acc[i] = (v8f){0, 0, 0, 0, 0, 0, 0, 0};

#pragma unroll 1
  for (int chunk = 0; chunk < 13; ++chunk) {
    __syncthreads();
    {  // stage chunk: 2 threads per row, 64 cols each
      int row = t >> 1;
      int cb = (t & 1) * 64;
      int node = n0 + row;
      unsigned* q = reinterpret_cast<unsigned*>(Ab + row * HH + cb);
      if (node < NN) {
        if (chunk == 0) {
          const unsigned* p =
              reinterpret_cast<const unsigned*>(hbf + (size_t)node * HH + cb);
#pragma unroll 8
          for (int j = 0; j < 32; ++j) q[j] = p[j];
        } else {
          int qsel = (chunk - 1) & 3;
          int sc = (chunk - 1) >> 2;
          float has = hin[node];
          float scale = (sc == 0) ? 1.0f : ((sc == 1) ? amp[node] : att[node]);
          float sdi = sdinv[node];
          size_t base = (size_t)node * HH + cb;
          for (int cc = 0; cc < 64; cc += 2) {
            float v0 = 0.f, v1 = 0.f;
            if (has > 0.5f) {
              v0 = aggval(qsel, s1, s2, mxu, mnu, base + cc, sdi) * scale;
              v1 = aggval(qsel, s1, s2, mxu, mnu, base + cc + 1, sdi) * scale;
            }
            q[cc >> 1] = packbf(v0, v1);
          }
        }
      } else {
#pragma unroll 8
        for (int j = 0; j < 32; ++j) q[j] = 0u;
      }
    }
    __syncthreads();

#pragma unroll
    for (int kt = 0; kt < 4; ++kt) {
      Frag16 a;
      const u32x4* ap = reinterpret_cast<const u32x4*>(
          Ab + (wv * 16 + ln) * HH + kt * 32 + hl * 8);
      a.q[0] = ap[0]; a.q[1] = ap[2];
#pragma unroll
      for (int nt = 0; nt < 8; ++nt) {
        Frag16 b;
        const u32x4* bp = reinterpret_cast<const u32x4*>(
            pw + ((size_t)((chunk * 4 + kt) * 8 + nt)) * 512 + lane * 16);
        b.q[0] = bp[0]; b.q[1] = bp[1];
        acc[nt] = wmma_bf16(a.v, b.v, acc[nt]);
      }
    }
  }

#pragma unroll
  for (int nt = 0; nt < 8; ++nt) {
    int col = nt * 16 + ln;
    float bc = bias[col];
#pragma unroll
    for (int r = 0; r < 8; ++r) {
      int node = n0 + wv * 16 + hl * 8 + r;
      if (node < NN) {
        size_t idx = (size_t)node * HH + col;
        float v = acc[nt][r] + bc + h[idx];   // residual
        h[idx] = v;
        hbf[idx] = (unsigned short)f2bfbits(v);
      }
    }
  }
}

// ---------------------------------------------------------------------------
// Readout: graphs are contiguous runs of 25 nodes -> [min | max | mean]
// ---------------------------------------------------------------------------
__global__ void k_readout(const float* __restrict__ h, float* __restrict__ r) {
  int g = blockIdx.x;
  int c = threadIdx.x;
  const float* p = h + (size_t)g * NPG * HH + c;
  float mn = p[0], mx = p[0], sm = p[0];
  for (int i = 1; i < NPG; ++i) {
    float v = p[(size_t)i * HH];
    mn = fminf(mn, v); mx = fmaxf(mx, v); sm += v;
  }
  r[(size_t)g * 384 + c] = mn;
  r[(size_t)g * 384 + 128 + c] = mx;
  r[(size_t)g * 384 + 256 + c] = sm * (1.0f / NPG);
}

// latent3d = r3 @ w_out3 + b_out3   ([G,384]x[384,256])
__global__ __launch_bounds__(128) void k_latent(
    const float* __restrict__ r3, const unsigned short* __restrict__ pw,
    const float* __restrict__ bias, float* __restrict__ lat) {
  __shared__ __align__(16) unsigned short Ab[32 * 384];
  const int t = threadIdx.x;
  const int g0 = blockIdx.x * 32;
  for (int i = t; i < 32 * 192; i += 128) {
    int row = i / 192;
    int c = (i - row * 192) * 2;
    int g = g0 + row;
    unsigned v = 0u;
    if (g < GG) v = packbf(r3[(size_t)g * 384 + c], r3[(size_t)g * 384 + c + 1]);
    reinterpret_cast<unsigned*>(Ab)[i] = v;
  }
  __syncthreads();
  const int wv = t >> 5, lane = t & 31, hl = lane >> 4, ln = lane & 15;
  const int rowt = wv & 1, ch = wv >> 1;   // ch selects cols 0-127 / 128-255
  v8f acc[8];
#pragma unroll
  for (int i = 0; i < 8; ++i) acc[i] = (v8f){0, 0, 0, 0, 0, 0, 0, 0};
#pragma unroll 1
  for (int kt = 0; kt < 12; ++kt) {
    Frag16 a;
    const u32x4* ap = reinterpret_cast<const u32x4*>(
        Ab + (rowt * 16 + ln) * 384 + kt * 32 + hl * 8);
    a.q[0] = ap[0]; a.q[1] = ap[2];
#pragma unroll
    for (int nt = 0; nt < 8; ++nt) {
      Frag16 b;
      const u32x4* bp = reinterpret_cast<const u32x4*>(
          pw + ((size_t)(kt * 16 + ch * 8 + nt)) * 512 + lane * 16);
      b.q[0] = bp[0]; b.q[1] = bp[1];
      acc[nt] = wmma_bf16(a.v, b.v, acc[nt]);
    }
  }
#pragma unroll
  for (int nt = 0; nt < 8; ++nt) {
    int col = ch * 128 + nt * 16 + ln;
    float bc = bias[col];
#pragma unroll
    for (int r = 0; r < 8; ++r) {
      int g = g0 + rowt * 16 + hl * 8 + r;
      if (g < GG) lat[(size_t)g * 256 + col] = acc[nt][r] + bc;
    }
  }
}

// out = relu([r2, latent] @ w1 + b1) @ w2 + b2
__global__ __launch_bounds__(128) void k_final(
    const float* __restrict__ r2, const float* __restrict__ lat,
    const unsigned short* __restrict__ pw1, const float* __restrict__ b1,
    const unsigned short* __restrict__ pw2, const float* __restrict__ b2,
    float* __restrict__ out) {
  __shared__ __align__(16) unsigned short Ab[32 * 640];
  __shared__ __align__(16) unsigned short H1[32 * 128];
  const int t = threadIdx.x;
  const int g0 = blockIdx.x * 32;
  for (int i = t; i < 32 * 320; i += 128) {
    int row = i / 320;
    int c = (i - row * 320) * 2;
    int g = g0 + row;
    unsigned v = 0u;
    if (g < GG) {
      float f0, f1;
      if (c < 384) { f0 = r2[(size_t)g * 384 + c]; f1 = r2[(size_t)g * 384 + c + 1]; }
      else { f0 = lat[(size_t)g * 256 + (c - 384)]; f1 = lat[(size_t)g * 256 + (c - 383)]; }
      v = packbf(f0, f1);
    }
    reinterpret_cast<unsigned*>(Ab)[i] = v;
  }
  __syncthreads();
  const int wv = t >> 5, lane = t & 31, hl = lane >> 4, ln = lane & 15;
  const int rowt = wv >> 1, ch = wv & 1;   // ch selects 64-col half
  v8f acc[4];
#pragma unroll
  for (int i = 0; i < 4; ++i) acc[i] = (v8f){0, 0, 0, 0, 0, 0, 0, 0};
#pragma unroll 1
  for (int kt = 0; kt < 20; ++kt) {
    Frag16 a;
    const u32x4* ap = reinterpret_cast<const u32x4*>(
        Ab + (rowt * 16 + ln) * 640 + kt * 32 + hl * 8);
    a.q[0] = ap[0]; a.q[1] = ap[2];
#pragma unroll
    for (int nt = 0; nt < 4; ++nt) {
      Frag16 b;
      const u32x4* bp = reinterpret_cast<const u32x4*>(
          pw1 + ((size_t)(kt * 8 + ch * 4 + nt)) * 512 + lane * 16);
      b.q[0] = bp[0]; b.q[1] = bp[1];
      acc[nt] = wmma_bf16(a.v, b.v, acc[nt]);
    }
  }
#pragma unroll
  for (int nt = 0; nt < 4; ++nt) {
    int col = ch * 64 + nt * 16 + ln;
    float bb = b1[col];
#pragma unroll
    for (int r = 0; r < 8; ++r) {
      int m = rowt * 16 + hl * 8 + r;
      float v = fmaxf(acc[nt][r] + bb, 0.0f);   // relu
      H1[m * 128 + col] = (unsigned short)f2bfbits(v);
    }
  }
  __syncthreads();
  v8f acc2[4];
#pragma unroll
  for (int i = 0; i < 4; ++i) acc2[i] = (v8f){0, 0, 0, 0, 0, 0, 0, 0};
#pragma unroll
  for (int kt = 0; kt < 4; ++kt) {
    Frag16 a;
    const u32x4* ap = reinterpret_cast<const u32x4*>(
        H1 + (rowt * 16 + ln) * 128 + kt * 32 + hl * 8);
    a.q[0] = ap[0]; a.q[1] = ap[2];
#pragma unroll
    for (int nt = 0; nt < 4; ++nt) {
      Frag16 b;
      const u32x4* bp = reinterpret_cast<const u32x4*>(
          pw2 + ((size_t)(kt * 8 + ch * 4 + nt)) * 512 + lane * 16);
      b.q[0] = bp[0]; b.q[1] = bp[1];
      acc2[nt] = wmma_bf16(a.v, b.v, acc2[nt]);
    }
  }
#pragma unroll
  for (int nt = 0; nt < 4; ++nt) {
    int col = ch * 64 + nt * 16 + ln;
    float bb = b2[col];
#pragma unroll
    for (int r = 0; r < 8; ++r) {
      int g = g0 + rowt * 16 + hl * 8 + r;
      if (g < GG) out[(size_t)g * 128 + col] = acc2[nt][r] + bb;
    }
  }
}

// ---------------------------------------------------------------------------
// Host launcher
// ---------------------------------------------------------------------------
extern "C" void kernel_launch(void* const* d_in, const int* in_sizes, int n_in,
                              void* d_out, int out_size, void* d_ws, size_t ws_size,
                              hipStream_t stream) {
  (void)in_sizes; (void)n_in; (void)out_size; (void)ws_size;
  const int* node_feat = (const int*)d_in[0];
  const int* edge_feat = (const int*)d_in[1];
  const int* srcI = (const int*)d_in[2];
  const int* dstI = (const int*)d_in[3];
  const float* aemb_f = (const float*)d_in[5];
  const float* bemb_f = (const float*)d_in[6];
  const float* w_pre_f = (const float*)d_in[7];
  const float* b_pre_f = (const float*)d_in[8];
  const float* w_post_f = (const float*)d_in[9];
  const float* b_post_f = (const float*)d_in[10];
  const float* w_out3 = (const float*)d_in[11];
  const float* b_out3 = (const float*)d_in[12];
  const float* aembT = (const float*)d_in[13];
  const float* bembT = (const float*)d_in[14];
  const float* w_preT = (const float*)d_in[15];
  const float* b_preT = (const float*)d_in[16];
  const float* w_postT = (const float*)d_in[17];
  const float* b_postT = (const float*)d_in[18];
  const float* w1 = (const float*)d_in[19];
  const float* b1 = (const float*)d_in[20];
  const float* w2 = (const float*)d_in[21];
  const float* b2 = (const float*)d_in[22];

  char* base = (char*)d_ws;
  size_t off = 0;
  auto alloc = [&](size_t bytes) -> char* {
    char* p = base + off;
    off = (off + bytes + 255) & ~(size_t)255;
    return p;
  };
  unsigned* degU = (unsigned*)alloc((size_t)NN * 4);
  float* sdinv = (float*)alloc((size_t)NN * 4);
  float* ampv = (float*)alloc((size_t)NN * 4);
  float* attv = (float*)alloc((size_t)NN * 4);
  float* hinv = (float*)alloc((size_t)NN * 4);
  float* h = (float*)alloc((size_t)NN * HH * 4);
  unsigned short* hbf = (unsigned short*)alloc((size_t)NN * HH * 2);
  float* s1 = (float*)alloc((size_t)NN * HH * 4);
  float* s2 = (float*)alloc((size_t)NN * HH * 4);
  unsigned* mxu = (unsigned*)alloc((size_t)NN * HH * 4);
  unsigned* mnu = (unsigned*)alloc((size_t)NN * HH * 4);
  float* r3 = (float*)alloc((size_t)GG * 384 * 4);
  float* r2 = (float*)alloc((size_t)GG * 384 * 4);
  float* lat = (float*)alloc((size_t)GG * 256 * 4);
  unsigned short* pWpre = (unsigned short*)alloc((size_t)10 * 384 * HH * 2);
  unsigned short* pWpost = (unsigned short*)alloc((size_t)10 * 1664 * HH * 2);
  unsigned short* pWout3 = (unsigned short*)alloc((size_t)384 * 256 * 2);
  unsigned short* pW1 = (unsigned short*)alloc((size_t)640 * HH * 2);
  unsigned short* pW2 = (unsigned short*)alloc((size_t)HH * HH * 2);

  // degree + PNA scalers (shared by both branches)
  k_zero_deg<<<(NN + 255) / 256, 256, 0, stream>>>(degU);
  k_count<<<(EE + 255) / 256, 256, 0, stream>>>(dstI, degU);
  k_scalars<<<(NN + 255) / 256, 256, 0, stream>>>(degU, sdinv, ampv, attv, hinv);

  // pack all weights into WMMA B-fragment bf16 layout
  for (int l = 0; l < 5; ++l) {
    k_pack<<<(384 * HH + 255) / 256, 256, 0, stream>>>(
        w_pre_f + (size_t)l * 384 * HH, pWpre + (size_t)l * 384 * HH, 384, HH);
    k_pack<<<(384 * HH + 255) / 256, 256, 0, stream>>>(
        w_preT + (size_t)l * 384 * HH, pWpre + (size_t)(5 + l) * 384 * HH, 384, HH);
    k_pack<<<(1664 * HH + 255) / 256, 256, 0, stream>>>(
        w_post_f + (size_t)l * 1664 * HH, pWpost + (size_t)l * 1664 * HH, 1664, HH);
    k_pack<<<(1664 * HH + 255) / 256, 256, 0, stream>>>(
        w_postT + (size_t)l * 1664 * HH, pWpost + (size_t)(5 + l) * 1664 * HH, 1664, HH);
  }
  k_pack<<<(384 * 256 + 255) / 256, 256, 0, stream>>>(w_out3, pWout3, 384, 256);
  k_pack<<<(640 * HH + 255) / 256, 256, 0, stream>>>(w1, pW1, 640, HH);
  k_pack<<<(HH * HH + 255) / 256, 256, 0, stream>>>(w2, pW2, HH, HH);

  for (int br = 0; br < 2; ++br) {
    const float* ae = br ? aembT : aemb_f;
    const float* be = br ? bembT : bemb_f;
    const float* bp = br ? b_preT : b_pre_f;
    const float* bq = br ? b_postT : b_post_f;
    const unsigned short* pwp = pWpre + (size_t)(br ? 5 : 0) * 384 * HH;
    const unsigned short* pwq = pWpost + (size_t)(br ? 5 : 0) * 1664 * HH;
    float* rout = br ? r2 : r3;

    k_embed<<<(NN * 64 + 127) / 128, 128, 0, stream>>>(node_feat, ae, h, hbf);
    for (int l = 0; l < 5; ++l) {
      k_init_aggr<<<(NN * HH + 255) / 256, 256, 0, stream>>>(s1, s2, mxu, mnu);
      k_pretrans<<<EE / 64, 128, 0, stream>>>(
          hbf, be, edge_feat, srcI, dstI,
          pwp + (size_t)l * 384 * HH, bp + (size_t)l * HH, s1, s2, mxu, mnu);
      k_posttrans<<<(NN + 63) / 64, 128, 0, stream>>>(
          h, hbf, s1, s2, mxu, mnu, sdinv, ampv, attv, hinv,
          pwq + (size_t)l * 1664 * HH, bq + (size_t)l * HH);
    }
    k_readout<<<GG, 128, 0, stream>>>(h, rout);
    if (br == 0)
      k_latent<<<(GG + 31) / 32, 128, 0, stream>>>(r3, pWout3, b_out3, lat);
  }
  k_final<<<(GG + 31) / 32, 128, 0, stream>>>(r2, lat, pW1, b1, pW2, b2,
                                              (float*)d_out);
}
